// ActorCritic_1176821039443
// MI455X (gfx1250) — matrix operand order repaired and checked by reading, not compiled, not run on hardware
//
#include <hip/hip_runtime.h>
#include <stdint.h>

#define BB   32
#define NN   512
#define FIN  20
#define DD   128
#define LL   5
#define NEGV (-10000000.0f)

typedef __attribute__((ext_vector_type(16))) __bf16 v16bf;
typedef __attribute__((ext_vector_type(8)))  float  v8f;

union Frag { v16bf v; uint4 q[2]; };

#if defined(__HIP_DEVICE_COMPILE__) && defined(__gfx1250__) && \
    __has_builtin(__builtin_amdgcn_global_load_async_to_lds_b128) && \
    __has_builtin(__builtin_amdgcn_s_wait_asynccnt)
#define USE_ASYNC 1
#else
#define USE_ASYNC 0
#endif

#if USE_ASYNC
typedef int vi4 __attribute__((vector_size(16)));   // matches builtin's expected v4i* params
__device__ __forceinline__ void async16(const unsigned short* g, unsigned short* l) {
    // async DMA: 16 bytes per lane, global -> LDS, tracked by ASYNCcnt
    __builtin_amdgcn_global_load_async_to_lds_b128((vi4*)g, (vi4*)l, 0, 0);
}
__device__ __forceinline__ void async_wait0() { __builtin_amdgcn_s_wait_asynccnt(0); }
#endif

__device__ __forceinline__ unsigned short f2bf(float f) {
    unsigned u = __builtin_bit_cast(unsigned, f);
    if ((u & 0x7f800000u) == 0x7f800000u) return (unsigned short)(u >> 16); // inf/nan: truncate
    unsigned r = u + 0x7fffu + ((u >> 16) & 1u);                            // RNE
    return (unsigned short)(r >> 16);
}

// ---------------- prep: adj (B,N,N,3) f32 -> adjT (3,B,N,N) bf16 ----------------
__global__ void k_prep_adj(const float* __restrict__ adj, unsigned short* __restrict__ adjT) {
    int t = blockIdx.x * 256 + threadIdx.x;      // one (b,n,m) per thread
    int b = t >> 18;
    int rem = t & (262144 - 1);
    int n = rem >> 9, m = rem & 511;
#pragma unroll
    for (int r = 0; r < 3; ++r)
        adjT[(((size_t)(r * BB + b) * NN + n) << 9) + m] = f2bf(adj[(size_t)t * 3 + r]);
}

// ------------- prep: actor_W/critic_W (L,3,D,D) f32 -> bf16 transposed [l][r][e][d] -------------
__global__ void k_prep_w(const float* __restrict__ aW, const float* __restrict__ cW,
                         unsigned short* __restrict__ aT, unsigned short* __restrict__ cT) {
    int idx = blockIdx.x * 256 + threadIdx.x;    // 0 .. 2*L*3*D*D-1
    const float* src; unsigned short* dst; int i;
    if (idx < LL * 3 * DD * DD) { src = aW; dst = aT; i = idx; }
    else                        { src = cW; dst = cT; i = idx - LL * 3 * DD * DD; }
    int lr = i / (DD * DD);
    int rem = i % (DD * DD);
    int d = rem / DD, e = rem % DD;
    dst[lr * DD * DD + e * DD + d] = f2bf(src[i]);
}

// ---------------- embed: h = relu(fatoms @ W) for actor (y=0) / critic (y=1) ----------------
__global__ void k_embed(const float* __restrict__ fatoms, const float* __restrict__ Wa,
                        const float* __restrict__ Wc, float* __restrict__ hA, float* __restrict__ hC) {
    __shared__ float fr[FIN];
    int row = blockIdx.x, j = threadIdx.x;
    const float* W = blockIdx.y ? Wc : Wa;
    float* h = blockIdx.y ? hC : hA;
    if (j < FIN) fr[j] = fatoms[row * FIN + j];
    __syncthreads();
    float a = 0.f;
#pragma unroll
    for (int k = 0; k < FIN; ++k) a += fr[k] * W[k * DD + j];
    h[(size_t)row * DD + j] = fmaxf(a, 0.f);
}

// ---------------- hw[r] = h @ W[r] + b[r], stored transposed bf16: hwT[r][b][e][m] ----------------
// block: 256 thr = 8 waves; 32 rows x 128 cols; WMMA bf16 16x16x32, K=128 in 4 steps
__global__ void k_hw(const float* __restrict__ h, const unsigned short* __restrict__ Wt,
                     const float* __restrict__ bias, unsigned short* __restrict__ hwT) {
    __shared__ __attribute__((aligned(16))) unsigned short Ah[32 * DD];   // 8 KB
    __shared__ __attribute__((aligned(16))) unsigned short Wl[DD * DD];   // 32 KB
    int r  = blockIdx.y;
    int g0 = blockIdx.x * 32;                    // global row base in [0, B*N)
    const unsigned short* Wr = Wt + r * DD * DD; // [e][d] bf16
#if USE_ASYNC
    // async DMA the 32KB weight tile while the VALU converts activations below
    for (int i = threadIdx.x; i < DD * DD / 8; i += 256)
        async16(Wr + i * 8, Wl + i * 8);
#else
    {
        const uint4* s = (const uint4*)Wr; uint4* dq = (uint4*)Wl;
        for (int i = threadIdx.x; i < DD * DD / 8; i += 256) dq[i] = s[i];
    }
#endif
    for (int i = threadIdx.x; i < 32 * DD; i += 256)
        Ah[i] = f2bf(h[(size_t)g0 * DD + i]);
#if USE_ASYNC
    async_wait0();
#endif
    __syncthreads();

    int w = threadIdx.x >> 5, lane = threadIdx.x & 31;
    int nw = w >> 1, mw = w & 1;
    int half = lane >> 4, lm = lane & 15;

    v8f acc[2] = {};
    for (int kk = 0; kk < 4; ++kk) {
        int k0 = kk * 32;
        Frag a;
        const unsigned short* ap = Ah + (mw * 16 + lm) * DD + k0 + half * 8;
        a.q[0] = *(const uint4*)ap;
        a.q[1] = *(const uint4*)(ap + 16);
#pragma unroll
        for (int t = 0; t < 2; ++t) {
            int n = nw * 32 + t * 16 + lm;
            const unsigned short* bp = Wl + n * DD + k0 + half * 8;
            Frag bfr;
            bfr.q[0] = *(const uint4*)bp;
            bfr.q[1] = *(const uint4*)(bp + 16);
            acc[t] = __builtin_amdgcn_wmma_f32_16x16x32_bf16(false, a.v, false, bfr.v,
                                                             (short)0, acc[t], false, false);
        }
    }
    int b = g0 >> 9;
    int mbase = (g0 & 511) + mw * 16 + half * 8;
#pragma unroll
    for (int t = 0; t < 2; ++t) {
        int n = nw * 32 + t * 16 + lm;
        float bv = bias[r * DD + n];
#pragma unroll
        for (int v = 0; v < 8; ++v) {
            int m = mbase + v;
            hwT[(((size_t)(r * BB + b) * DD + n) << 9) + m] = f2bf(acc[t][v] + bv);
        }
    }
}

// ---------------- h_out[b] = mask * sum_r relu(adjT[r][b] @ hw[r][b]) ----------------
// block: 256 thr = 8 waves (4 m-waves x 2 n-waves); tile 64 rows x 128 cols
// K=512 in 16 steps x 3 relations = 48 staged tiles, double-buffered async DMA
__global__ void k_msg(const unsigned short* __restrict__ adjT, const unsigned short* __restrict__ hwT,
                      const float* __restrict__ mask, float* __restrict__ out) {
    __shared__ __attribute__((aligned(16))) unsigned short At[2][64 * 32];   // 2 x 4 KB
    __shared__ __attribute__((aligned(16))) unsigned short Bt[2][DD * 32];   // 2 x 8 KB
    int b  = blockIdx.x;
    int m0 = blockIdx.y * 64;
    int tid = threadIdx.x;
    int w = tid >> 5, lane = tid & 31;
    int mw = w & 3, nw = w >> 2;
    int half = lane >> 4, lm = lane & 15;
    int arow = tid >> 2, apart = tid & 3;        // A staging: 1 x b128 per thread

    auto issue = [&](int it) {
        int r = it >> 4, kk = it & 15;
        int k0 = kk * 32;
        int p = it & 1;
        const unsigned short* Abase = adjT + (((size_t)(r * BB + b) * NN + m0) << 9);
        const unsigned short* Bbase = hwT + (((size_t)(r * BB + b) * DD) << 9);
#if USE_ASYNC
        async16(Abase + ((size_t)arow << 9) + k0 + apart * 8, &At[p][tid * 8]);
#pragma unroll
        for (int jj = 0; jj < 2; ++jj) {
            int j = tid + jj * 256;
            int br = j >> 2, bp = j & 3;
            async16(Bbase + ((size_t)br << 9) + k0 + bp * 8, &Bt[p][j * 8]);
        }
#else
        ((uint4*)At[p])[tid] = *(const uint4*)(Abase + ((size_t)arow << 9) + k0 + apart * 8);
#pragma unroll
        for (int jj = 0; jj < 2; ++jj) {
            int j = tid + jj * 256;
            int br = j >> 2, bp = j & 3;
            ((uint4*)Bt[p])[j] = *(const uint4*)(Bbase + ((size_t)br << 9) + k0 + bp * 8);
        }
#endif
    };

    v8f accS[4] = {};
    v8f accR[4] = {};
#if USE_ASYNC
    issue(0);                                    // prologue: DMA tile 0
#endif
    for (int it = 0; it < 48; ++it) {
        int p = it & 1;
#if USE_ASYNC
        async_wait0();                           // my tile-`it` copies have landed
        __syncthreads();                         // all waves landed; all done reading buf p^1 from it-1
        if (it + 1 < 48) issue(it + 1);          // DMA next tile, overlapped with WMMA below
#else
        __syncthreads();
        issue(it);
        __syncthreads();
#endif
        Frag a;
        const unsigned short* ap = At[p] + (mw * 16 + lm) * 32 + half * 8;
        a.q[0] = *(const uint4*)ap;
        a.q[1] = *(const uint4*)(ap + 16);
#pragma unroll
        for (int t = 0; t < 4; ++t) {
            int n = nw * 64 + t * 16 + lm;
            const unsigned short* bpp = Bt[p] + n * 32 + half * 8;
            Frag bfr;
            bfr.q[0] = *(const uint4*)bpp;
            bfr.q[1] = *(const uint4*)(bpp + 16);
            accR[t] = __builtin_amdgcn_wmma_f32_16x16x32_bf16(false, a.v, false, bfr.v,
                                                              (short)0, accR[t], false, false);
        }
        if ((it & 15) == 15) {                   // end of one relation: fold ReLU, reset
#pragma unroll
            for (int t = 0; t < 4; ++t) {
#pragma unroll
                for (int v = 0; v < 8; ++v) {
                    accS[t][v] += fmaxf(accR[t][v], 0.f);
                    accR[t][v] = 0.f;
                }
            }
        }
    }
    int mb = m0 + mw * 16 + half * 8;
#pragma unroll
    for (int v = 0; v < 8; ++v) {
        int m = mb + v;
        float mk = mask[b * NN + m];
#pragma unroll
        for (int t = 0; t < 4; ++t) {
            int n = nw * 64 + t * 16 + lm;
            out[(((size_t)b * NN + m) << 7) + n] = accS[t][v] * mk;
        }
    }
}

// ---------------- critic pooling: sum & masked max over n ----------------
__global__ void k_pool(const float* __restrict__ hC, const float* __restrict__ cmask,
                       float* __restrict__ qin) {
    int b = blockIdx.x, e = threadIdx.x;
    float s = 0.f, mx = -3.4e38f;
    for (int n = 0; n < NN; ++n) {
        float v = hC[((size_t)b * NN + n) * DD + e];
        float cm = cmask[b * NN + n];
        s += v * cm;
        mx = fmaxf(mx, v + (cm == 1.f ? 0.f : NEGV));
    }
    qin[b * 2 * DD + e] = s;
    qin[b * 2 * DD + DD + e] = mx;
}

// ---------------- qvalue head ----------------
__global__ void k_q(const float* __restrict__ qin, const float* __restrict__ Wq1,
                    const float* __restrict__ bq1, const float* __restrict__ Wq2,
                    const float* __restrict__ bq2, float* __restrict__ out) {
    __shared__ float red[DD];
    int b = blockIdx.x, j = threadIdx.x;
    float a = bq1[j];
    for (int k = 0; k < 2 * DD; ++k) a += qin[b * 2 * DD + k] * Wq1[k * DD + j];
    red[j] = fmaxf(a, 0.f) * Wq2[j];
    __syncthreads();
    for (int s = 64; s > 0; s >>= 1) { if (j < s) red[j] += red[j + s]; __syncthreads(); }
    if (j == 0) out[128 + b] = red[0] + bq2[0];
}

// ---------------- gather focus/end rows of actor output ----------------
__global__ void k_gather(const float* __restrict__ hA, const int* __restrict__ focus,
                         const int* __restrict__ act, float* __restrict__ startv,
                         float* __restrict__ endv) {
    int b = blockIdx.x, e = threadIdx.x;
    int f  = focus[b];
    int a0 = act[b * 3 + 0];
    startv[b * DD + e] = hA[((size_t)b * NN + f) * DD + e];
    endv[b * DD + e]   = hA[((size_t)b * NN + a0) * DD + e];
}

// ---------------- end logits: relu([start, x_a] @ We1 + be1) @ We2 + pm ----------------
__global__ void k_endlog(const float* __restrict__ startv, const float* __restrict__ hA,
                         const float* __restrict__ We1, const float* __restrict__ be1,
                         const float* __restrict__ We2, const float* __restrict__ pmask,
                         float* __restrict__ logits) {
    __shared__ float red[DD];
    int n = blockIdx.x, b = blockIdx.y, j = threadIdx.x;
    const float* hr = hA + ((size_t)b * NN + n) * DD;
    const float* st = startv + b * DD;
    float a = be1[j];
    for (int k = 0; k < DD; ++k) a += st[k] * We1[k * DD + j];
    for (int k = 0; k < DD; ++k) a += hr[k] * We1[(DD + k) * DD + j];
    red[j] = fmaxf(a, 0.f) * We2[j];
    __syncthreads();
    for (int s = 64; s > 0; s >>= 1) { if (j < s) red[j] += red[j + s]; __syncthreads(); }
    if (j == 0) {
        float pm = (pmask[b * NN + n] == 1.f) ? 0.f : NEGV;
        logits[b * NN + n] = red[0] + pm;
    }
}

// ---------------- softmax over N + entropy + end_prob gather ----------------
__global__ void k_softmax(const float* __restrict__ logits, const int* __restrict__ act,
                          float* __restrict__ out, float* __restrict__ entEnd) {
    __shared__ float red[256];
    __shared__ float Ms, Ss;
    int b = blockIdx.x, t = threadIdx.x;
    float l0 = logits[b * NN + t], l1 = logits[b * NN + 256 + t];
    red[t] = fmaxf(l0, l1);
    __syncthreads();
    for (int s = 128; s > 0; s >>= 1) { if (t < s) red[t] = fmaxf(red[t], red[t + s]); __syncthreads(); }
    if (t == 0) Ms = red[0];
    __syncthreads();
    float M = Ms;
    float e0 = expf(l0 - M), e1 = expf(l1 - M);
    red[t] = e0 + e1;
    __syncthreads();
    for (int s = 128; s > 0; s >>= 1) { if (t < s) red[t] += red[t + s]; __syncthreads(); }
    if (t == 0) Ss = red[0];
    __syncthreads();
    float S = Ss;
    red[t] = e0 * (l0 - M) + e1 * (l1 - M);
    __syncthreads();
    for (int s = 128; s > 0; s >>= 1) { if (t < s) red[t] += red[t + s]; __syncthreads(); }
    if (t == 0) {
        float T = red[0];
        entEnd[b] = logf(S) - T / S;               // -sum p log p
        int ei = act[b * 3 + 0];
        out[b * 3 + 0] = expf(logits[b * NN + ei] - M) / S;
    }
}

// ---------------- bond + stop heads, final entropy ----------------
__global__ void k_bondstop(const float* __restrict__ startv, const float* __restrict__ endv,
                           const float* __restrict__ Wb1, const float* __restrict__ bb1,
                           const float* __restrict__ Wb2, const float* __restrict__ Ws1,
                           const float* __restrict__ bs1, const float* __restrict__ Ws2,
                           const int* __restrict__ act, const float* __restrict__ entEnd,
                           float* __restrict__ out) {
    __shared__ float hid[DD];
    __shared__ float red[DD];
    __shared__ float lg[5];
    int b = blockIdx.x, j = threadIdx.x;
    const float* st = startv + b * DD;
    const float* en = endv + b * DD;
    float a = bb1[j];
    for (int k = 0; k < DD; ++k) a += st[k] * Wb1[k * DD + j];
    for (int k = 0; k < DD; ++k) a += en[k] * Wb1[(DD + k) * DD + j];
    hid[j] = fmaxf(a, 0.f);
    __syncthreads();
    for (int c = 0; c < 3; ++c) {
        red[j] = hid[j] * Wb2[j * 3 + c];
        __syncthreads();
        for (int s = 64; s > 0; s >>= 1) { if (j < s) red[j] += red[j + s]; __syncthreads(); }
        if (j == 0) lg[c] = red[0];
        __syncthreads();
    }
    float a2 = bs1[j];
    for (int k = 0; k < DD; ++k) a2 += st[k] * Ws1[k * DD + j];
    float h2 = fmaxf(a2, 0.f);
    for (int c = 0; c < 2; ++c) {
        red[j] = h2 * Ws2[j * 2 + c];
        __syncthreads();
        for (int s = 64; s > 0; s >>= 1) { if (j < s) red[j] += red[j + s]; __syncthreads(); }
        if (j == 0) lg[3 + c] = red[0];
        __syncthreads();
    }
    if (j == 0) {
        float m3 = fmaxf(lg[0], fmaxf(lg[1], lg[2]));
        float b0 = expf(lg[0] - m3), b1 = expf(lg[1] - m3), b2 = expf(lg[2] - m3);
        float sb = b0 + b1 + b2;
        float p0 = b0 / sb, p1 = b1 / sb, p2 = b2 / sb;
        float entB = -(p0 * logf(p0) + p1 * logf(p1) + p2 * logf(p2));
        int a1 = act[b * 3 + 1];
        float m2 = fmaxf(lg[3], lg[4]);
        float s0 = expf(lg[3] - m2), s1 = expf(lg[4] - m2);
        float ss = s0 + s1;
        float q0 = s0 / ss, q1 = s1 / ss;
        float entS = -(q0 * logf(q0) + q1 * logf(q1));
        int a2i = act[b * 3 + 2];
        out[b * 3 + 1] = (a1 == 0) ? p0 : (a1 == 1 ? p1 : p2);
        out[b * 3 + 2] = (a2i == 0) ? q0 : q1;
        out[96 + b] = entEnd[b] + entB + entS;
    }
}

extern "C" void kernel_launch(void* const* d_in, const int* in_sizes, int n_in,
                              void* d_out, int out_size, void* d_ws, size_t ws_size,
                              hipStream_t stream) {
    (void)in_sizes; (void)n_in; (void)out_size; (void)ws_size;
    const float* fatoms       = (const float*)d_in[0];
    const float* adj          = (const float*)d_in[1];
    const float* all_mask     = (const float*)d_in[2];
    const float* current_mask = (const float*)d_in[3];
    const float* prob_mask    = (const float*)d_in[4];
    const int*   focus        = (const int*)d_in[5];
    const int*   act          = (const int*)d_in[6];
    const float* W_emb        = (const float*)d_in[7];
    const float* W_cemb       = (const float*)d_in[8];
    const float* actor_W      = (const float*)d_in[9];
    const float* actor_b      = (const float*)d_in[10];
    const float* critic_W     = (const float*)d_in[11];
    const float* critic_b     = (const float*)d_in[12];
    const float* We1 = (const float*)d_in[13];
    const float* be1 = (const float*)d_in[14];
    const float* We2 = (const float*)d_in[15];
    const float* Wb1 = (const float*)d_in[16];
    const float* bb1 = (const float*)d_in[17];
    const float* Wb2 = (const float*)d_in[18];
    const float* Ws1 = (const float*)d_in[19];
    const float* bs1 = (const float*)d_in[20];
    const float* Ws2 = (const float*)d_in[21];
    const float* Wq1 = (const float*)d_in[22];
    const float* bq1 = (const float*)d_in[23];
    const float* Wq2 = (const float*)d_in[24];
    const float* bq2 = (const float*)d_in[25];
    float* out = (float*)d_out;

    char* ws = (char*)d_ws;
    size_t off = 0;
    auto alloc = [&](size_t bytes) -> void* {
        void* p = ws + off; off += (bytes + 255) & ~(size_t)255; return p;
    };
    unsigned short* adjT = (unsigned short*)alloc((size_t)3 * BB * NN * NN * 2);
    unsigned short* WbfA = (unsigned short*)alloc((size_t)LL * 3 * DD * DD * 2);
    unsigned short* WbfC = (unsigned short*)alloc((size_t)LL * 3 * DD * DD * 2);
    float* hA0 = (float*)alloc((size_t)BB * NN * DD * 4);
    float* hA1 = (float*)alloc((size_t)BB * NN * DD * 4);
    float* hC0 = (float*)alloc((size_t)BB * NN * DD * 4);
    float* hC1 = (float*)alloc((size_t)BB * NN * DD * 4);
    unsigned short* hwT = (unsigned short*)alloc((size_t)3 * BB * DD * NN * 2);
    float* qin    = (float*)alloc((size_t)BB * 2 * DD * 4);
    float* startv = (float*)alloc((size_t)BB * DD * 4);
    float* endv   = (float*)alloc((size_t)BB * DD * 4);
    float* logits = (float*)alloc((size_t)BB * NN * 4);
    float* entEnd = (float*)alloc((size_t)BB * 4);

    k_prep_adj<<<(BB * NN * NN) / 256, 256, 0, stream>>>(adj, adjT);
    k_prep_w<<<(2 * LL * 3 * DD * DD) / 256, 256, 0, stream>>>(actor_W, critic_W, WbfA, WbfC);
    k_embed<<<dim3(BB * NN, 2), DD, 0, stream>>>(fatoms, W_emb, W_cemb, hA0, hC0);

    float* ha_in = hA0; float* ha_out = hA1;
    float* hc_in = hC0; float* hc_out = hC1;
    for (int l = 0; l < LL; ++l) {
        k_hw<<<dim3(BB * NN / 32, 3), 256, 0, stream>>>(ha_in, WbfA + l * 3 * DD * DD,
                                                        actor_b + l * 3 * DD, hwT);
        k_msg<<<dim3(BB, NN / 64), 256, 0, stream>>>(adjT, hwT, all_mask, ha_out);
        { float* t = ha_in; ha_in = ha_out; ha_out = t; }
        k_hw<<<dim3(BB * NN / 32, 3), 256, 0, stream>>>(hc_in, WbfC + l * 3 * DD * DD,
                                                        critic_b + l * 3 * DD, hwT);
        k_msg<<<dim3(BB, NN / 64), 256, 0, stream>>>(adjT, hwT, all_mask, hc_out);
        { float* t = hc_in; hc_in = hc_out; hc_out = t; }
    }

    k_pool<<<BB, DD, 0, stream>>>(hc_in, current_mask, qin);
    k_q<<<BB, DD, 0, stream>>>(qin, Wq1, bq1, Wq2, bq2, out);
    k_gather<<<BB, DD, 0, stream>>>(ha_in, focus, act, startv, endv);
    k_endlog<<<dim3(NN, BB), DD, 0, stream>>>(startv, ha_in, We1, be1, We2, prob_mask, logits);
    k_softmax<<<BB, 256, 0, stream>>>(logits, act, out, entEnd);
    k_bondstop<<<BB, DD, 0, stream>>>(startv, endv, Wb1, bb1, Wb2, Ws1, bs1, Ws2, act, entEnd, out);
}
